// ScaledDotProduct_67542655696923
// MI455X (gfx1250) — matrix operand order repaired and checked by reading, not compiled, not run on hardware
//
#include <hip/hip_runtime.h>
#include <hip/hip_bf16.h>

typedef _Float16 half16 __attribute__((ext_vector_type(16)));
typedef _Float16 half2t __attribute__((ext_vector_type(2)));
typedef float    float8 __attribute__((ext_vector_type(8)));

#define QTILE 128
#define KTILE 64
#define DDIM  128
#define DPAD  136   // 128 + 8 halfs pad -> conflict-free ds strides
#define VPAD  72    // 64 + 8 halfs pad
#define SMAX  2048

// Flash-attention with ghostmax (softmax-with-extra-0-logit), causal + additive
// attention mask. f16 WMMA (v_wmma_f32_16x16x32_f16) with f32 accumulation.
//  - softmax in exp2 domain (log2e folded into Q scale / mask constants),
//    raw v_exp_f32 via __builtin_amdgcn_exp2f
//  - row sums of P computed by WMMA against an all-ones B fragment (replaces
//    the ds_bpermute sum-reduction chains)
//  - packed (float8) accumulator rescale and denominator update
// One block = 8 waves = 128 query rows of one (b,h); each wave owns 16 rows.
__global__ __launch_bounds__(256)
void fa_ghostmax_kernel(const float* __restrict__ Q,
                        const float* __restrict__ K,
                        const float* __restrict__ V,
                        const int*   __restrict__ amask,
                        const int*   __restrict__ causal_p,
                        float* __restrict__ O,
                        int S, int H)
{
  __shared__ _Float16 Qs[QTILE][DPAD];   // [q row][d], scaled by log2e/sqrt(D)
  __shared__ _Float16 Ks[KTILE][DPAD];   // [key][d]
  __shared__ _Float16 Vt[DDIM][VPAD];    // [d][key]  (transposed V tile)
  __shared__ _Float16 Ps[8][16][VPAD];   // per-wave P scratch for C->A transpose
  __shared__ float    AmL[SMAX];         // additive mask term, log2 domain

  const int tid  = threadIdx.x;
  const int wave = tid >> 5;
  const int lane = tid & 31;
  const int ln   = lane & 15;   // N column / row-within-16
  const int hh   = lane >> 4;   // lane half

  const int b  = blockIdx.z;
  const int h  = blockIdx.y;
  const int q0 = blockIdx.x * QTILE;
  const long long base = ((long long)(b * H + h)) * S * DDIM;
  const int causal = causal_p[0];
  // (1/sqrt(128)) * log2(e): scores come out of WMMA already in log2 domain
  const float scl = 0.08838834764831845f * 1.4426950408889634f;

  // ---- stage Q tile (pre-scaled) into LDS as f16; stage mask terms ----
  {
    const float4* src = (const float4*)(Q + base + (long long)q0 * DDIM);
    #pragma unroll
    for (int it = 0; it < (QTILE * DDIM / 4) / 256; ++it) {
      int i = tid + it * 256;
      float4 f = src[i];
      int r = i >> 5, c = (i & 31) << 2;
      Qs[r][c+0] = (_Float16)(f.x * scl);
      Qs[r][c+1] = (_Float16)(f.y * scl);
      Qs[r][c+2] = (_Float16)(f.z * scl);
      Qs[r][c+3] = (_Float16)(f.w * scl);
    }
    for (int i = tid; i < S; i += 256)  // (1 - am) * (-1e12 * log2e)
      AmL[i] = (1.0f - (float)amask[b * S + i]) * -1.4426950408889634e12f;
  }
  __syncthreads();

  // ---- per-wave Q A-fragments: 4 chunks of K=32 over D=128 ----
  // 16-bit A 16x32 layout: lane%16 = M row; kb(v,half) = (v<4?2v:16+2(v-4)) + 8*half
  half16 qf[4];
  {
    const int row = wave * 16 + ln;
    #pragma unroll
    for (int dc = 0; dc < 4; ++dc)
      #pragma unroll
      for (int v = 0; v < 8; ++v) {
        int kb = ((v < 4) ? 2*v : 16 + 2*(v-4)) + 8*hh;
        half2t p = *(const half2t*)&Qs[row][dc*32 + kb];
        qf[dc][2*v]   = p[0];
        qf[dc][2*v+1] = p[1];
      }
  }

  // all-ones B fragment: D = P x ones  ->  every lane holds its row's sum(P)
  half16 onesf;
  #pragma unroll
  for (int i = 0; i < 16; ++i) onesf[i] = (_Float16)1.0f;

  // 16x128 f32 output accumulator as 8 C-fragments; C row = vgpr + 8*half
  float8 acc[8];
  #pragma unroll
  for (int nc = 0; nc < 8; ++nc)
    #pragma unroll
    for (int j = 0; j < 8; ++j) acc[nc][j] = 0.0f;

  float m[8];
  float8 lv;
  #pragma unroll
  for (int r = 0; r < 8; ++r) { m[r] = 0.0f; lv[r] = 1.0f; }  // ghost logit 0

  const int qrow0 = q0 + wave * 16;
  const int kend  = causal ? (q0 + QTILE < S ? q0 + QTILE : S) : S;

  for (int ks = 0; ks < kend; ks += KTILE) {
    __syncthreads();  // prior iteration's LDS reads done before overwrite
    // ---- stage K tile and transposed V tile ----
    {
      const float4* ksrc = (const float4*)(K + base + (long long)ks * DDIM);
      const float4* vsrc = (const float4*)(V + base + (long long)ks * DDIM);
      #pragma unroll
      for (int it = 0; it < (KTILE * DDIM / 4) / 256; ++it) {
        int i = tid + it * 256;
        int r = i >> 5, c = (i & 31) << 2;
        float4 f = ksrc[i];
        Ks[r][c+0] = (_Float16)f.x;
        Ks[r][c+1] = (_Float16)f.y;
        Ks[r][c+2] = (_Float16)f.z;
        Ks[r][c+3] = (_Float16)f.w;
        float4 g = vsrc[i];
        Vt[c+0][r] = (_Float16)g.x;
        Vt[c+1][r] = (_Float16)g.y;
        Vt[c+2][r] = (_Float16)g.z;
        Vt[c+3][r] = (_Float16)g.w;
      }
    }
    __syncthreads();

    // ---- S = Q * K^T : 16 q-rows x 64 keys (four 16x16 C-frags) ----
    // B 32x16 layout: lane%16 = N (key); K rows = (v<4?2v:16+2(v-4)) + 8*half + {0,1}
    float8 sf[4];
    #pragma unroll
    for (int j = 0; j < 4; ++j) {
      float8 c;
      #pragma unroll
      for (int e = 0; e < 8; ++e) c[e] = 0.0f;
      #pragma unroll
      for (int dc = 0; dc < 4; ++dc) {
        half16 bf;
        #pragma unroll
        for (int v = 0; v < 8; ++v) {
          int d = dc*32 + ((v < 4) ? 2*v : 16 + 2*(v-4)) + 8*hh;
          half2t p = *(const half2t*)&Ks[j*16 + ln][d];
          bf[2*v]   = p[0];
          bf[2*v+1] = p[1];
        }
        c = __builtin_amdgcn_wmma_f32_16x16x32_f16(false, qf[dc], false, bf,
                                                   (short)0, c, false, false);
      }
      sf[j] = c;
    }

    // ---- online ghost-softmax update (log2 domain) ----
    float amadd[4];
    #pragma unroll
    for (int j = 0; j < 4; ++j) amadd[j] = AmL[ks + j*16 + ln];
    // wave-uniform: does this key tile touch the wave's diagonal?
    const bool needCausal = causal && (ks + KTILE - 1 > qrow0);

    float8 scv;  // per-row accumulator rescale factors
    #pragma unroll
    for (int r = 0; r < 8; ++r) {
      const int qi = qrow0 + r + 8*hh;
      float s[4];
      #pragma unroll
      for (int j = 0; j < 4; ++j) s[j] = sf[j][r] + amadd[j];
      if (needCausal) {
        #pragma unroll
        for (int j = 0; j < 4; ++j)
          if (ks + j*16 + ln > qi) s[j] = -1e30f;
      }
      float mx = fmaxf(fmaxf(s[0], s[1]), fmaxf(s[2], s[3]));
      mx = fmaxf(mx, __shfl_xor(mx, 1));   // row reduce across 16 lanes
      mx = fmaxf(mx, __shfl_xor(mx, 2));
      mx = fmaxf(mx, __shfl_xor(mx, 4));
      mx = fmaxf(mx, __shfl_xor(mx, 8));
      float mn = fmaxf(m[r], mx);          // m starts at 0 (ghost logit)
      scv[r] = __builtin_amdgcn_exp2f(m[r] - mn);
      m[r] = mn;
      float p0 = __builtin_amdgcn_exp2f(s[0] - mn);
      float p1 = __builtin_amdgcn_exp2f(s[1] - mn);
      float p2 = __builtin_amdgcn_exp2f(s[2] - mn);
      float p3 = __builtin_amdgcn_exp2f(s[3] - mn);
      const int row = r + 8*hh;            // C-layout -> scratch (transpose)
      Ps[wave][row][ln]      = (_Float16)p0;
      Ps[wave][row][16 + ln] = (_Float16)p1;
      Ps[wave][row][32 + ln] = (_Float16)p2;
      Ps[wave][row][48 + ln] = (_Float16)p3;
    }

    // packed rescale of output accumulators
    #pragma unroll
    for (int nc = 0; nc < 8; ++nc) acc[nc] *= scv;

    // ---- O += P * V ; rowsum(P) via ones-WMMA (two K=32 chunks) ----
    float8 sumf;
    #pragma unroll
    for (int e = 0; e < 8; ++e) sumf[e] = 0.0f;
    #pragma unroll
    for (int kc = 0; kc < 2; ++kc) {
      half16 pf;  // P as A-fragment 16x32 (same-wave DS ordering covers RAW)
      #pragma unroll
      for (int v = 0; v < 8; ++v) {
        int kb = ((v < 4) ? 2*v : 16 + 2*(v-4)) + 8*hh;
        half2t p = *(const half2t*)&Ps[wave][ln][kc*32 + kb];
        pf[2*v]   = p[0];
        pf[2*v+1] = p[1];
      }
      sumf = __builtin_amdgcn_wmma_f32_16x16x32_f16(false, pf, false, onesf,
                                                    (short)0, sumf, false, false);
      #pragma unroll
      for (int nc = 0; nc < 8; ++nc) {
        half16 bf;  // V as B 32x16: N = d column, K rows = keys
        #pragma unroll
        for (int v = 0; v < 8; ++v) {
          int key = kc*32 + ((v < 4) ? 2*v : 16 + 2*(v-4)) + 8*hh;
          half2t p = *(const half2t*)&Vt[nc*16 + ln][key];
          bf[2*v]   = p[0];
          bf[2*v+1] = p[1];
        }
        acc[nc] = __builtin_amdgcn_wmma_f32_16x16x32_f16(false, pf, false, bf,
                                                         (short)0, acc[nc],
                                                         false, false);
      }
    }
    // packed denominator update: l = l*sc + rowsum
    lv = lv * scv + sumf;
  }

  // ---- epilogue: divide by ghostmax denominator, store f32 ----
  float8 invv;
  #pragma unroll
  for (int r = 0; r < 8; ++r) invv[r] = 1.0f / lv[r];
  #pragma unroll
  for (int nc = 0; nc < 8; ++nc)
    #pragma unroll
    for (int r = 0; r < 8; ++r) {
      int row = qrow0 + r + 8*hh;
      O[base + (long long)row * DDIM + nc*16 + ln] = acc[nc][r] * invv[r];
    }
}

extern "C" void kernel_launch(void* const* d_in, const int* in_sizes, int n_in,
                              void* d_out, int out_size, void* d_ws, size_t ws_size,
                              hipStream_t stream) {
  const float* q  = (const float*)d_in[0];
  const float* k  = (const float*)d_in[1];
  const float* v  = (const float*)d_in[2];
  const int*   am = (const int*)d_in[3];
  const int*   causal = (const int*)d_in[6];   // read on device
  float* out = (float*)d_out;

  // Fixed problem shape from the reference setup: B=2, H=16, S=2048, D=128.
  const int H = 16, D = 128, S = 2048;
  const int B = in_sizes[0] / (H * S * D);

  dim3 grid(S / QTILE, H, B);
  fa_ghostmax_kernel<<<grid, 256, 0, stream>>>(q, k, v, am, causal, out, S, H);

  (void)n_in; (void)out_size; (void)d_ws; (void)ws_size;
}
